// Head_11098195493114
// MI455X (gfx1250) — matrix operand order repaired and checked by reading, not compiled, not run on hardware
//
#include <hip/hip_runtime.h>

typedef _Float16 h16;
typedef __attribute__((ext_vector_type(16))) _Float16 v16h;
typedef __attribute__((ext_vector_type(8)))  float    v8f;
typedef __attribute__((ext_vector_type(4)))  unsigned u32x4;
typedef __attribute__((ext_vector_type(8)))  int      i32x8;
typedef __attribute__((ext_vector_type(4)))  int      i32x4;

#if __has_builtin(__builtin_amdgcn_tensor_load_to_lds)
#define HAVE_TDM 1
#if __has_include(<hip/amd_detail/amd_gfx1250_TDM.h>)
#define TDM_SIX_ARGS 1
#endif
#endif

static constexpr int BB = 8;       // batch
static constexpr int TT = 2048;    // sequence
static constexpr int CC = 1024;    // embed
static constexpr int DD = 64;      // head size
static constexpr int ROWS = BB * TT;

__device__ __forceinline__ v8f wmma16(v16h a, v16h b, v8f c) {
  // v_wmma_f32_16x16x32_f16 : D = A(16x32 f16) * B(32x16 f16) + C(f32)
  return __builtin_amdgcn_wmma_f32_16x16x32_f16(false, a, false, b, (short)0, c,
                                                false, false);
}

__device__ __forceinline__ v8f zero8() {
  v8f z;
#pragma unroll
  for (int i = 0; i < 8; ++i) z[i] = 0.0f;
  return z;
}

// A-matrix (16x32 f16) K index for element e of a lane: one contiguous
// 16-half chunk at K = 2*a0+0..15, a0 = (lane>=16)*8.
__device__ __forceinline__ int amapK(int e, int a0) {
  return 2 * a0 + e;
}

// ---------------------------------------------------------------------------
// Kernel 0: transpose + hi/lo split weights: W[1024][64] f32 -> WT[3][64][1024]
// ---------------------------------------------------------------------------
__global__ void prep_w(const float* __restrict__ Wq, const float* __restrict__ Wk,
                       const float* __restrict__ Wv, h16* __restrict__ wh,
                       h16* __restrict__ wl) {
  int idx = blockIdx.x * blockDim.x + threadIdx.x;
  if (idx >= 3 * DD * CC) return;
  int m = idx / (DD * CC);
  int r = idx % (DD * CC);
  int n = r / CC;
  int k = r % CC;
  const float* W = (m == 0) ? Wq : ((m == 1) ? Wk : Wv);
  float v = W[(size_t)k * DD + n];
  h16 h = (h16)v;
  wh[idx] = h;
  wl[idx] = (h16)(v - (float)h);
}

// ---------------------------------------------------------------------------
// Kernel 1: QKV projection. Block = 256 thr = 8 waves; wave (rt,nt) owns the
// 16x16 tile (rows blockIdx.x*32 + rt*16, cols nt*16) of matrix blockIdx.y.
// K=1024 contraction, hi/lo split -> 3 WMMAs per 32-step.
// Front-end: TDM tensor_load_to_lds stages a 32x128 fp32 x-tile in LDS
// (TENSORcnt + workgroup barrier), shared by all 8 waves.
// q,k stored [B*T][64] (hi/lo); v stored transposed [B][64][T] (hi/lo).
// ---------------------------------------------------------------------------
__global__ __launch_bounds__(256) void qkv_proj(
    const float* __restrict__ x, const h16* __restrict__ wh,
    const h16* __restrict__ wl, h16* __restrict__ qh, h16* __restrict__ ql,
    h16* __restrict__ kh, h16* __restrict__ kl, h16* __restrict__ vh,
    h16* __restrict__ vl) {
#if HAVE_TDM
  __shared__ float xtile[32 * 128];  // 16 KB staging tile
#endif
  const int lane = threadIdx.x & 31;
  const int wave = threadIdx.x >> 5;
  const int mat = blockIdx.y;  // 0=q 1=k 2=v
  const int rt = wave >> 2;
  const int nt = wave & 3;
  const int row0 = blockIdx.x * 32 + rt * 16;
  const int ln = lane & 15;
  const int a0 = (lane >= 16) ? 8 : 0;
  const int b0 = (lane >= 16) ? 16 : 0;
  const int ncol = nt * 16 + ln;

  const h16* whp = wh + ((size_t)mat * DD + ncol) * CC;
  const h16* wlp = wl + ((size_t)mat * DD + ncol) * CC;

  v8f acc = zero8();

#if HAVE_TDM
  for (int kc = 0; kc < CC; kc += 128) {
    if (wave == 0) {
      // Build 2-D tensor descriptor (D#) per CDNA5 ISA ch.8:
      // tile = 128 (dim0) x 32 rows (dim1) of 4-byte elements.
      unsigned long long ga =
          (unsigned long long)(const void*)x +
          ((unsigned long long)(blockIdx.x * 32) * CC + (unsigned)kc) * 4ull;
      unsigned ldsoff = (unsigned)(unsigned long long)(void*)xtile;
      u32x4 g0 = {1u,                                   // count=1 (valid D#)
                  ldsoff,                               // lds_addr
                  (unsigned)(ga & 0xFFFFFFFFu),         // global_addr[31:0]
                  (unsigned)((ga >> 32) & 0x1FFFFFFu) | (2u << 30)};  // type=2
      i32x8 g1 = {(int)(2u << 16),                      // data_size = 4B
                  (int)((unsigned)(CC & 0xFFFF) << 16), // tensor_dim0 lo
                  (int)(((unsigned)CC >> 16) |
                        ((unsigned)(ROWS & 0xFFFF) << 16)),  // dim0 hi | dim1 lo
                  (int)(((unsigned)ROWS >> 16) | (128u << 16)),  // dim1 hi | tile_dim0
                  (int)32,                              // tile_dim1 = 32 rows
                  (int)CC,                              // tensor_dim0_stride lo
                  0, 0};
      i32x4 gz = {0, 0, 0, 0};
#if TDM_SIX_ARGS
      i32x8 gz8 = {0, 0, 0, 0, 0, 0, 0, 0};
      __builtin_amdgcn_tensor_load_to_lds(g0, g1, gz, gz, gz8, 0);
#else
      __builtin_amdgcn_tensor_load_to_lds(g0, g1, gz, gz, 0);
#endif
      __builtin_amdgcn_s_wait_tensorcnt(0);
    }
    __syncthreads();  // publish LDS tile to all waves
#pragma unroll
    for (int sub = 0; sub < 4; ++sub) {
      const int ks = kc + sub * 32;
      const float* xs = &xtile[(rt * 16 + ln) * 128 + sub * 32];
      v16h ah, al, bh, bl;
#pragma unroll
      for (int e = 0; e < 16; ++e) {
        float xv = xs[amapK(e, a0)];
        h16 h = (h16)xv;
        ah[e] = h;
        al[e] = (h16)(xv - (float)h);
      }
#pragma unroll
      for (int e = 0; e < 16; ++e) {
        bh[e] = whp[ks + b0 + e];
        bl[e] = wlp[ks + b0 + e];
      }
      acc = wmma16(ah, bh, acc);
      acc = wmma16(ah, bl, acc);
      acc = wmma16(al, bh, acc);
    }
    __syncthreads();  // all reads done before tile is overwritten
  }
#else
  const float* xr = x + (size_t)(row0 + ln) * CC;
  for (int ks = 0; ks < CC; ks += 32) {
    v16h ah, al, bh, bl;
#pragma unroll
    for (int e = 0; e < 16; ++e) {
      float xv = xr[ks + amapK(e, a0)];
      h16 h = (h16)xv;
      ah[e] = h;
      al[e] = (h16)(xv - (float)h);
    }
#pragma unroll
    for (int e = 0; e < 16; ++e) {
      bh[e] = whp[ks + b0 + e];
      bl[e] = wlp[ks + b0 + e];
    }
    acc = wmma16(ah, bh, acc);
    acc = wmma16(ah, bl, acc);
    acc = wmma16(al, bh, acc);
  }
#endif

  const int mb = (lane >= 16) ? 8 : 0;
#pragma unroll
  for (int r = 0; r < 8; ++r) {
    int mrow = row0 + r + mb;
    float v = acc[r];
    h16 h = (h16)v;
    h16 lo = (h16)(v - (float)h);
    if (mat == 0) {
      qh[(size_t)mrow * DD + ncol] = h;
      ql[(size_t)mrow * DD + ncol] = lo;
    } else if (mat == 1) {
      kh[(size_t)mrow * DD + ncol] = h;
      kl[(size_t)mrow * DD + ncol] = lo;
    } else {
      int bat = mrow >> 11, t = mrow & (TT - 1);
      size_t vi = ((size_t)bat * DD + ncol) * TT + t;
      vh[vi] = h;
      vl[vi] = lo;
    }
  }
}

// ---------------------------------------------------------------------------
// Kernel 2: fused causal flash-attention. One 16-row query tile per wave,
// online softmax over 32-key steps. 12 WMMAs for q*k^T (+split), LDS bounce
// for the P C-layout -> A-fragment transpose (s_wait_dscnt guarded),
// 12 WMMAs for p*v (+split). global_prefetch_b8 hides next-step K/V latency.
// ---------------------------------------------------------------------------
__global__ __launch_bounds__(256) void attn(
    const h16* __restrict__ qh, const h16* __restrict__ ql,
    const h16* __restrict__ kh, const h16* __restrict__ kl,
    const h16* __restrict__ vh, const h16* __restrict__ vl,
    float* __restrict__ out) {
  __shared__ float pscr[8][16 * 32];  // per-wave P transpose scratch (16 KB)

  const int lane = threadIdx.x & 31;
  const int wave = threadIdx.x >> 5;
  const int tile = blockIdx.x * 8 + wave;  // 0..1023
  const int bat = tile >> 7;               // batch
  const int qi = (tile & 127) * 16;        // query tile start
  const int ln = lane & 15;
  const int a0 = (lane >= 16) ? 8 : 0;
  const int b0 = (lane >= 16) ? 16 : 0;
  const int mb = (lane >= 16) ? 8 : 0;
  float* scr = pscr[wave];

  // Preload Q A-fragments (2 d-steps, hi/lo) — loop invariant.
  const h16* qhr = qh + (size_t)(bat * TT + qi + ln) * DD;
  const h16* qlr = ql + (size_t)(bat * TT + qi + ln) * DD;
  v16h qah[2], qal[2];
#pragma unroll
  for (int s = 0; s < 2; ++s)
#pragma unroll
    for (int e = 0; e < 16; ++e) {
      int K = s * 32 + amapK(e, a0);
      qah[s][e] = qhr[K];
      qal[s][e] = qlr[K];
    }

  v8f o[4];
#pragma unroll
  for (int n = 0; n < 4; ++n) o[n] = zero8();
  float mrow[8], lrow[8];
#pragma unroll
  for (int r = 0; r < 8; ++r) {
    mrow[r] = -1e30f;
    lrow[r] = 0.0f;
  }

  const h16* kbh = kh + (size_t)bat * TT * DD;
  const h16* kbl = kl + (size_t)bat * TT * DD;
  const h16* vbh = vh + (size_t)bat * DD * TT;
  const h16* vbl = vl + (size_t)bat * DD * TT;

  const int jmask0 = (qi >> 5) * 32;  // first step needing the causal mask
  const int jend = qi + 15;

  for (int j = 0; j <= jend; j += 32) {
    const bool masked = (j >= jmask0);

    // Prefetch next step's K/V rows into cache (global_prefetch_b8).
    {
      int jn = j + 32;
      if (jn <= jend) {
        __builtin_prefetch(kbh + (size_t)(jn + ln) * DD, 0, 3);
        __builtin_prefetch(kbh + (size_t)(jn + 16 + ln) * DD, 0, 3);
        __builtin_prefetch(kbl + (size_t)(jn + ln) * DD, 0, 3);
        __builtin_prefetch(vbh + (size_t)ln * TT + jn, 0, 3);
      }
    }

    // ---- scores: two 16x16 tiles over 32 keys, K=64 contraction ----
    v8f s[2];
#pragma unroll
    for (int t = 0; t < 2; ++t) {
      s[t] = zero8();
      const h16* krh = kbh + (size_t)(j + t * 16 + ln) * DD;
      const h16* krl = kbl + (size_t)(j + t * 16 + ln) * DD;
#pragma unroll
      for (int ds = 0; ds < 2; ++ds) {
        v16h bh, bl;
#pragma unroll
        for (int e = 0; e < 16; ++e) {
          bh[e] = krh[ds * 32 + b0 + e];
          bl[e] = krl[ds * 32 + b0 + e];
        }
        s[t] = wmma16(qah[ds], bh, s[t]);
        s[t] = wmma16(qah[ds], bl, s[t]);
        s[t] = wmma16(qal[ds], bh, s[t]);
      }
    }

    // ---- scale + causal mask ----
#pragma unroll
    for (int t = 0; t < 2; ++t)
#pragma unroll
      for (int r = 0; r < 8; ++r) {
        float sv = s[t][r] * 0.125f;  // 1/sqrt(64)
        if (masked) {
          int kcol = j + t * 16 + ln;
          int qrow = qi + r + mb;
          if (kcol > qrow) sv = -1e30f;
        }
        s[t][r] = sv;
      }

    // ---- online softmax: row stats via intra-half xor shuffles ----
    float cf[8];
#pragma unroll
    for (int r = 0; r < 8; ++r) {
      float v = fmaxf(s[0][r], s[1][r]);
#pragma unroll
      for (int m = 1; m < 16; m <<= 1) v = fmaxf(v, __shfl_xor(v, m, 32));
      float mn = fmaxf(mrow[r], v);
      cf[r] = __expf(mrow[r] - mn);
      mrow[r] = mn;
    }
#pragma unroll
    for (int r = 0; r < 8; ++r) {
      float p0 = __expf(s[0][r] - mrow[r]);
      float p1 = __expf(s[1][r] - mrow[r]);
      s[0][r] = p0;
      s[1][r] = p1;
      float v = p0 + p1;
#pragma unroll
      for (int m = 1; m < 16; m <<= 1) v += __shfl_xor(v, m, 32);
      lrow[r] = lrow[r] * cf[r] + v;
    }
#pragma unroll
    for (int n = 0; n < 4; ++n)
#pragma unroll
      for (int r = 0; r < 8; ++r) o[n][r] *= cf[r];

    // ---- P: C-layout -> A-fragment via per-wave LDS scratch ----
#pragma unroll
    for (int t = 0; t < 2; ++t)
#pragma unroll
      for (int r = 0; r < 8; ++r) scr[(r + mb) * 32 + t * 16 + ln] = s[t][r];
    asm volatile("s_wait_dscnt 0" ::: "memory");  // DS in-order per wave
    v16h pah, pal;
#pragma unroll
    for (int e = 0; e < 16; ++e) {
      float pv = scr[ln * 32 + amapK(e, a0)];
      h16 h = (h16)pv;
      pah[e] = h;
      pal[e] = (h16)(pv - (float)h);
    }

    // ---- O += P * V (V transposed: rows of vT are contiguous in k) ----
#pragma unroll
    for (int n = 0; n < 4; ++n) {
      const h16* vrh = vbh + (size_t)(n * 16 + ln) * TT + j;
      const h16* vrl = vbl + (size_t)(n * 16 + ln) * TT + j;
      v16h bh, bl;
#pragma unroll
      for (int e = 0; e < 16; ++e) {
        bh[e] = vrh[b0 + e];
        bl[e] = vrl[b0 + e];
      }
      o[n] = wmma16(pah, bh, o[n]);
      o[n] = wmma16(pah, bl, o[n]);
      o[n] = wmma16(pal, bh, o[n]);
    }
  }

  // ---- epilogue: normalize and store fp32 ----
#pragma unroll
  for (int n = 0; n < 4; ++n)
#pragma unroll
    for (int r = 0; r < 8; ++r) {
      int row = qi + r + mb;
      int col = n * 16 + ln;
      out[((size_t)(bat * TT + row)) * DD + col] = o[n][r] / lrow[r];
    }
}

// ---------------------------------------------------------------------------
extern "C" void kernel_launch(void* const* d_in, const int* in_sizes, int n_in,
                              void* d_out, int out_size, void* d_ws,
                              size_t ws_size, hipStream_t stream) {
  const float* x = (const float*)d_in[0];
  const float* Wq = (const float*)d_in[1];
  const float* Wk = (const float*)d_in[2];
  const float* Wv = (const float*)d_in[3];
  float* out = (float*)d_out;

  size_t off = 0;
  auto carve = [&](size_t bytes) -> void* {
    void* p = (char*)d_ws + off;
    off += (bytes + 255) & ~(size_t)255;
    return p;
  };
  h16* wh = (h16*)carve((size_t)3 * DD * CC * sizeof(h16));
  h16* wl = (h16*)carve((size_t)3 * DD * CC * sizeof(h16));
  h16* qh = (h16*)carve((size_t)ROWS * DD * sizeof(h16));
  h16* ql = (h16*)carve((size_t)ROWS * DD * sizeof(h16));
  h16* kh = (h16*)carve((size_t)ROWS * DD * sizeof(h16));
  h16* kl = (h16*)carve((size_t)ROWS * DD * sizeof(h16));
  h16* vh = (h16*)carve((size_t)ROWS * DD * sizeof(h16));
  h16* vl = (h16*)carve((size_t)ROWS * DD * sizeof(h16));

  {
    int n = 3 * DD * CC;
    prep_w<<<(n + 255) / 256, 256, 0, stream>>>(Wq, Wk, Wv, wh, wl);
  }
  {
    dim3 grid(ROWS / 32, 3);
    qkv_proj<<<grid, 256, 0, stream>>>(x, wh, wl, qh, ql, kh, kl, vh, vl);
  }
  {
    int qtiles = BB * (TT / 16);  // 1024 tiles, 8 per block
    attn<<<qtiles / 8, 256, 0, stream>>>(qh, ql, kh, kl, vh, vl, out);
  }
}
